// NILUT_57827439674099
// MI455X (gfx1250) — matrix operand order repaired and checked by reading, not compiled, not run on hardware
//
#include <hip/hip_runtime.h>
#include <hip/hip_bf16.h>
#include <math.h>

// ---------------------------------------------------------------------------
// NILUT per-pixel MLP on MI455X (gfx1250), WMMA f16 path.
//
// Roofline: 524288 pixels x ~149K MACs = ~156 GFLOP vs ~13 MB of HBM traffic
// (0.56 us at 23.3 TB/s) -> massively compute-bound -> V_WMMA_F32_16X16X32_F16.
//
// Each wave (wave32) owns a 32-pixel tile (two 16-row WMMA M-tiles) and runs
// the whole MLP on it:
//   [32x96] x [96x256]  (layer0, input padded 67->96)
//   [32x256]x [256x256] (layers 1,2)
//   [32x256]x [256x16]  (layer3, output padded 3->16)
// Every B fragment (weights) feeds TWO WMMAs (both M-tiles) -> 2x compute per
// byte of weight traffic. Activations: f16 in LDS, XOR-swizzled at 16B-chunk
// granularity to avoid bank conflicts. Weights pre-transposed to f16 Wt[n][k]
// in d_ws so a B fragment is two 16B reads per lane (WGP$/L2 hot).
// GELU/sigmoid are branch-free exp2+rcp forms (2 TRANS + 5 VALU ops) that
// co-execute with the XDL WMMA pipe, instead of ocml's branchy erff.
// ---------------------------------------------------------------------------

typedef __attribute__((ext_vector_type(8)))  _Float16 v8h;
typedef __attribute__((ext_vector_type(16))) _Float16 v16h;
typedef __attribute__((ext_vector_type(8)))  float    v8f;

#define HID       256
#define K0P       96     // padded layer-0 input dim (3 rgb + 64 emb + 29 pad)
#define IN_DIM    67
#define STYLE_DIM 64
#define TILE_M    32     // pixels per wave (two 16-row WMMA M-tiles)
#define WAVES     4
#define BLK_THREADS (WAVES * 32)
#define BLK_PIX   (TILE_M * WAVES)

// Swizzled LDS offset (in halves) for element (m, k) of a wave's [32 x 256]
// activation tile. 16B chunks (8 halves) are XOR-swizzled by row so the
// per-lane ds_load_b128 A-fragment reads (16 rows, same k) spread banks.
__device__ __forceinline__ int sw_idx(int m, int k) {
  int c = (k >> 3) ^ (m & 7);
  return m * HID + (c << 3) + (k & 7);
}

// Branch-free GELU (tanh form), in exp2 domain:
//   gelu(x) ~= x * sigmoid(1.5957691*x + 0.0713548*x^3)
//   sigmoid(t) = 1/(1 + exp2(-t*log2e))
__device__ __forceinline__ float gelu_fast(float x) {
  float x2 = x * x;
  float p  = __builtin_fmaf(x2, -0.1029518599f, -2.3021208829f);
  float e  = __builtin_amdgcn_exp2f(x * p);
  return x * __builtin_amdgcn_rcpf(1.0f + e);
}

__device__ __forceinline__ float sigmoid_fast(float x) {
  float e = __builtin_amdgcn_exp2f(x * -1.4426950409f);
  return __builtin_amdgcn_rcpf(1.0f + e);
}

// Load one 16x32 f16 A-fragment per the CDNA5 WMMA layout:
//   lane L<16  : row M=L, K = {k0..k0+7}   u {k0+16..k0+23}
//   lane L+16  : row M=L, K = {k0+8..k0+15} u {k0+24..k0+31}
// -> two contiguous 8-half (16B) LDS loads per lane. `mrow` selects which
// 16-row M-tile of the wave's 32-row block.
__device__ __forceinline__ v16h ld_afrag(const _Float16* act, int mrow,
                                         int hsel, int kc) {
  const int kLo = kc * 32 + hsel * 8;
  v8h lo = *(const v8h*)(act + sw_idx(mrow, kLo));
  v8h hi = *(const v8h*)(act + sw_idx(mrow, kLo + 16));
  return __builtin_shufflevector(lo, hi, 0, 1, 2, 3, 4, 5, 6, 7,
                                         8, 9, 10, 11, 12, 13, 14, 15);
}

// One hidden layer on a 32-row tile: act[32 x KCH*32] x Wt[256 x KCH*32],
// GELU, written back into the same LDS tile (all A reads hoisted first).
template <int KCH>
__device__ __forceinline__ void mlp_layer(_Float16* act,
                                          const _Float16* __restrict__ wt,
                                          const float* __restrict__ bias,
                                          int lrow, int hsel) {
  const int KDIM = KCH * 32;
  v16h a0[KCH], a1[KCH];
#pragma unroll
  for (int kc = 0; kc < KCH; ++kc) {
    a0[kc] = ld_afrag(act, lrow,      hsel, kc);
    a1[kc] = ld_afrag(act, lrow + 16, hsel, kc);
  }

#pragma unroll 1
  for (int nt = 0; nt < HID / 16; ++nt) {
    const int n = nt * 16 + lrow;
    const float bv = bias[n];
    v8f acc0, acc1;
#pragma unroll
    for (int r = 0; r < 8; ++r) { acc0[r] = bv; acc1[r] = bv; }

    // B fragment: lane L<16 -> column N=n, K={k0..k0+7} u {k0+16..k0+23};
    // lane L+16 -> same column, K={k0+8..k0+15} u {k0+24..k0+31}. Reused by both M-tiles.
    const _Float16* wrow = wt + n * KDIM + hsel * 8;
#pragma unroll
    for (int kc = 0; kc < KCH; ++kc) {
      v8h blo = *(const v8h*)(wrow + kc * 32);
      v8h bhi = *(const v8h*)(wrow + kc * 32 + 16);
      v16h bfr = __builtin_shufflevector(blo, bhi, 0, 1, 2, 3, 4, 5, 6, 7, 8, 9, 10, 11, 12, 13, 14, 15);
      acc0 = __builtin_amdgcn_wmma_f32_16x16x32_f16(
          false, a0[kc], false, bfr, (short)0, acc0, false, false);
      acc1 = __builtin_amdgcn_wmma_f32_16x16x32_f16(
          false, a1[kc], false, bfr, (short)0, acc1, false, false);
    }

    // D layout: VGPR r, lanes 0-15 -> M=r, lanes 16-31 -> M=r+8; N = lane&15.
#pragma unroll
    for (int r = 0; r < 8; ++r) {
      act[sw_idx(r + hsel * 8,      n)] = (_Float16)gelu_fast(acc0[r]);
      act[sw_idx(r + hsel * 8 + 16, n)] = (_Float16)gelu_fast(acc1[r]);
    }
  }
  // Phase fence: LDS is in-order per wave; make it explicit and stop the
  // compiler migrating next-layer loads above the stores.
  asm volatile("s_wait_dscnt 0x0" ::: "memory");
}

__global__ __launch_bounds__(BLK_THREADS) void nilut_mlp(
    const float* __restrict__ rgb, const int* __restrict__ style_idx,
    const float* __restrict__ emb,
    const float* __restrict__ b0, const float* __restrict__ b1,
    const float* __restrict__ b2, const float* __restrict__ b3,
    const _Float16* __restrict__ wt0, const _Float16* __restrict__ wt1,
    const _Float16* __restrict__ wt2, const _Float16* __restrict__ wt3,
    float* __restrict__ out, int npix, int hw) {
  __shared__ _Float16 smem[WAVES * TILE_M * HID];  // 64 KB, per-wave tiles

  const int lane = threadIdx.x & 31;
  const int wave = threadIdx.x >> 5;
  const int lrow = lane & 15;
  const int hsel = lane >> 4;  // 0: lanes 0-15, 1: lanes 16-31
  _Float16* act = smem + wave * (TILE_M * HID);
  const int p0 = blockIdx.x * BLK_PIX + wave * TILE_M;

  // ---- Assemble layer-0 input in LDS: k 0..2 = rgb (CHW), 3..66 = style
  //      embedding, 67..95 = zero pad. Per-wave cooperative fill.
  for (int idx = lane; idx < TILE_M * K0P; idx += 32) {
    const int m = idx / K0P;
    const int k = idx - m * K0P;
    const int p = p0 + m;
    const int pc = (p < npix) ? p : (npix - 1);
    const int bidx = pc / hw;
    const int pix = pc - bidx * hw;
    float v = 0.0f;
    if (k < 3)
      v = rgb[(bidx * 3 + k) * hw + pix];
    else if (k < IN_DIM)
      v = emb[style_idx[bidx] * STYLE_DIM + (k - 3)];
    if (p >= npix) v = 0.0f;
    act[sw_idx(m, k)] = (_Float16)v;
  }
  asm volatile("s_wait_dscnt 0x0" ::: "memory");

  // ---- 3 hidden layers (GELU), all WMMA.
  mlp_layer<3>(act, wt0, b0, lrow, hsel);   // 96  -> 256
  mlp_layer<8>(act, wt1, b1, lrow, hsel);   // 256 -> 256
  mlp_layer<8>(act, wt2, b2, lrow, hsel);   // 256 -> 256

  // ---- Final layer: 256 -> 3 (one 16-wide N tile, cols 3..15 zero-padded).
  v16h a0[8], a1[8];
#pragma unroll
  for (int kc = 0; kc < 8; ++kc) {
    a0[kc] = ld_afrag(act, lrow,      hsel, kc);
    a1[kc] = ld_afrag(act, lrow + 16, hsel, kc);
  }
  const float bv = (lrow < 3) ? b3[lrow] : 0.0f;
  v8f acc0, acc1;
#pragma unroll
  for (int r = 0; r < 8; ++r) { acc0[r] = bv; acc1[r] = bv; }
  const _Float16* wrow = wt3 + lrow * HID + hsel * 8;
#pragma unroll
  for (int kc = 0; kc < 8; ++kc) {
    v8h blo = *(const v8h*)(wrow + kc * 32);
    v8h bhi = *(const v8h*)(wrow + kc * 32 + 16);
    v16h bfr = __builtin_shufflevector(blo, bhi, 0, 1, 2, 3, 4, 5, 6, 7, 8, 9, 10, 11, 12, 13, 14, 15);
    acc0 = __builtin_amdgcn_wmma_f32_16x16x32_f16(
        false, a0[kc], false, bfr, (short)0, acc0, false, false);
    acc1 = __builtin_amdgcn_wmma_f32_16x16x32_f16(
        false, a1[kc], false, bfr, (short)0, acc1, false, false);
  }

  // Sigmoid + scatter to NCHW output. N = lrow is the channel; only 0..2 live.
  if (lrow < 3) {
#pragma unroll
    for (int r = 0; r < 8; ++r) {
#pragma unroll
      for (int t = 0; t < 2; ++t) {
        const int p = p0 + t * 16 + r + hsel * 8;
        const float v = (t == 0) ? acc0[r] : acc1[r];
        if (p < npix) {
          const int bidx = p / hw;
          const int pix = p - bidx * hw;
          out[(bidx * 3 + lrow) * hw + pix] = sigmoid_fast(v);
        }
      }
    }
  }
}

// Pre-transpose + f16-convert all weights into d_ws:
//   wt0[n*96 + k]  = W0[k][n] (k<67, else 0)   n<256, k<96
//   wt1[n*256 + k] = W1[k][n]                  n<256, k<256
//   wt2[n*256 + k] = W2[k][n]
//   wt3[n*256 + k] = W3[k][n] (n<3, else 0)    n<16,  k<256
__global__ __launch_bounds__(256) void prep_weights(
    const float* __restrict__ W0, const float* __restrict__ W1,
    const float* __restrict__ W2, const float* __restrict__ W3,
    _Float16* __restrict__ wt0, _Float16* __restrict__ wt1,
    _Float16* __restrict__ wt2, _Float16* __restrict__ wt3) {
  int i = blockIdx.x * 256 + threadIdx.x;
  if (i < 256 * 96) {
    int n = i / 96, k = i - n * 96;
    wt0[i] = (k < IN_DIM) ? (_Float16)W0[k * 256 + n] : (_Float16)0.0f;
    return;
  }
  i -= 256 * 96;
  if (i < 256 * 256) {
    int n = i >> 8, k = i & 255;
    wt1[i] = (_Float16)W1[k * 256 + n];
    return;
  }
  i -= 256 * 256;
  if (i < 256 * 256) {
    int n = i >> 8, k = i & 255;
    wt2[i] = (_Float16)W2[k * 256 + n];
    return;
  }
  i -= 256 * 256;
  if (i < 16 * 256) {
    int n = i >> 8, k = i & 255;
    wt3[i] = (n < 3) ? (_Float16)W3[k * 3 + n] : (_Float16)0.0f;
  }
}

extern "C" void kernel_launch(void* const* d_in, const int* in_sizes, int n_in,
                              void* d_out, int out_size, void* d_ws,
                              size_t ws_size, hipStream_t stream) {
  const float* rgb  = (const float*)d_in[0];
  const int*   sidx = (const int*)d_in[1];
  const float* emb  = (const float*)d_in[2];
  const float* W0 = (const float*)d_in[3];
  const float* b0 = (const float*)d_in[4];
  const float* W1 = (const float*)d_in[5];
  const float* b1 = (const float*)d_in[6];
  const float* W2 = (const float*)d_in[7];
  const float* b2 = (const float*)d_in[8];
  const float* W3 = (const float*)d_in[9];
  const float* b3 = (const float*)d_in[10];
  float* out = (float*)d_out;

  const int B    = in_sizes[1];
  const int hw   = in_sizes[0] / (3 * B);
  const int npix = B * hw;

  _Float16* wt0 = (_Float16*)d_ws;
  _Float16* wt1 = wt0 + 256 * 96;
  _Float16* wt2 = wt1 + 256 * 256;
  _Float16* wt3 = wt2 + 256 * 256;

  // Weight prep: 159744 elements, deterministic every call.
  const int prep_total = 256 * 96 + 2 * 256 * 256 + 16 * 256;
  prep_weights<<<(prep_total + 255) / 256, 256, 0, stream>>>(
      W0, W1, W2, W3, wt0, wt1, wt2, wt3);

  const int blocks = (npix + BLK_PIX - 1) / BLK_PIX;
  nilut_mlp<<<blocks, BLK_THREADS, 0, stream>>>(rgb, sidx, emb, b0, b1, b2, b3,
                                                wt0, wt1, wt2, wt3, out, npix,
                                                hw);
}
